// ShapeGenerator3D_87419764343298
// MI455X (gfx1250) — matrix operand order repaired and checked by reading, not compiled, not run on hardware
//
#include <hip/hip_runtime.h>
#include <hip/hip_bf16.h>
#include <math.h>

typedef __attribute__((ext_vector_type(16))) _Float16 v16h;
typedef __attribute__((ext_vector_type(8)))  _Float16 v8h;
typedef __attribute__((ext_vector_type(8)))  float    v8f;

__device__ __forceinline__ int imin(int a, int b) { return a < b ? a : b; }
__device__ __forceinline__ int imax(int a, int b) { return a > b ? a : b; }

// 16-bit A-matrix 16x32 k order used here (contiguous; not AMD's table order):
// element e -> K = (lane&16?16:0) + e   (one contiguous 16-half run, matches B)
__device__ __forceinline__ int a_k_of(int lane, int e) {
  int v = e >> 1, p = e & 1;
  return ((lane & 16) ? 16 : 0) + (v << 1) + p;
}

// ---------- prep: f32 -> f16 with K zero-padding (clamp+select, no branchy load) ----------
__global__ __launch_bounds__(256)
void cvt_pad_f16_kernel(const float* __restrict__ src, _Float16* __restrict__ dst,
                        int rows, int srcK, int srcStride, int dstK) {
  int idx = blockIdx.x * blockDim.x + threadIdx.x;
  if (idx >= rows * dstK) return;
  int r = idx / dstK, k = idx - r * dstK;
  float v = src[(size_t)r * srcStride + imin(k, srcK - 1)];
  dst[idx] = (_Float16)((k < srcK) ? v : 0.0f);
}

// ---------------- FC: H0[64,8192] = z[:, :200] @ fc_w^T + fc_b ----------------
__global__ __launch_bounds__(256)
void fc_wmma_kernel(const _Float16* __restrict__ Af, const _Float16* __restrict__ Bf,
                    const float* __restrict__ fcb, float* __restrict__ H0) {
  int wave = (blockIdx.x * blockDim.x + threadIdx.x) >> 5;  // 2048 waves
  int lane = threadIdx.x & 31;
  int nt = wave & 511;          // 512 N tiles
  int mt = wave >> 9;           // 4 M tiles
  int m0 = mt * 16, n0 = nt * 16;
  int aoff = (lane & 16) ? 16 : 0;
  const _Float16* ap = Af + (m0 + (lane & 15)) * 224;
  const _Float16* bp = Bf + (size_t)(n0 + (lane & 15)) * 224 + ((lane & 16) ? 16 : 0);
  v8f acc = {};
#pragma unroll
  for (int kc = 0; kc < 7; ++kc) {   // K = 200 padded to 224
    int kb = kc * 32;
    union { v16h v; v8h h[2]; } au;
    au.h[0] = *(const v8h*)(ap + kb + aoff);        // K = kb+aoff .. +8
    au.h[1] = *(const v8h*)(ap + kb + 8 + aoff);    // K = kb+8+aoff .. +8
    v16h b = *(const v16h*)(bp + kb);               // K = kb+off2 .. +16
    acc = __builtin_amdgcn_wmma_f32_16x16x32_f16(false, au.v, false, b, (short)0, acc,
                                                 false, false);
  }
#pragma unroll
  for (int r = 0; r < 8; ++r) {
    int mrow = m0 + r + ((lane & 16) ? 8 : 0);
    int ncol = n0 + (lane & 15);
    H0[mrow * 8192 + ncol] = acc[r] + fcb[ncol];
  }
}

// -------- BN train-mode stats: per channel over (64 batch) x (S spatial) ------
__global__ __launch_bounds__(256)
void bn_stats_kernel(const float* __restrict__ X, const float* __restrict__ g,
                     const float* __restrict__ bb, float* __restrict__ scale,
                     float* __restrict__ shift, int C, int S) {
  int c = blockIdx.x;
  __shared__ float ssum[256], ssq[256];
  float s = 0.f, q = 0.f;
  int total = 64 * S;
  for (int t = threadIdx.x; t < total; t += blockDim.x) {
    int n = t / S, sp = t - n * S;
    float v = X[(size_t)n * C * S + (size_t)c * S + sp];
    s += v; q += v * v;
  }
  ssum[threadIdx.x] = s; ssq[threadIdx.x] = q;
  __syncthreads();
  for (int o = 128; o > 0; o >>= 1) {
    if (threadIdx.x < o) {
      ssum[threadIdx.x] += ssum[threadIdx.x + o];
      ssq[threadIdx.x]  += ssq[threadIdx.x + o];
    }
    __syncthreads();
  }
  if (threadIdx.x == 0) {
    float mean = ssum[0] / (float)total;
    float var  = ssq[0] / (float)total - mean * mean;
    float sc   = g[c] * rsqrtf(var + 1e-5f);
    scale[c] = sc;
    shift[c] = bb[c] - mean * sc;
  }
}

// ---- prep: BN+ReLU activations as half2 (pw,pw+1) pairs, zero halo pad ----
// Xpair[n][ci][pd][ph][pw0], pd,ph in [0,IN+1], pw0 in [0,IN]; 4-byte aligned pairs
__global__ __launch_bounds__(256)
void act_pair_prep_kernel(const float* __restrict__ X, const float* __restrict__ scale,
                          const float* __restrict__ shift, unsigned int* __restrict__ Xpair,
                          int C, int IN) {
  int PDm = IN + 2, PW2 = IN + 1;
  int PPV = PDm * PDm * PW2;
  int idx = blockIdx.x * blockDim.x + threadIdx.x;
  if (idx >= 64 * C * PPV) return;
  int pidx = idx % PPV; int t = idx / PPV; int ci = t % C; int n = t / C;
  int pd = pidx / (PDm * PW2); int rem = pidx % (PDm * PW2);
  int ph = rem / PW2; int pw0 = rem % PW2;
  float sc = scale[ci], sh = shift[ci];
  const float* Xc = X + ((size_t)n * C + ci) * (IN * IN * IN);
  int d = pd - 1, h = ph - 1;
  bool inDH = ((unsigned)d < (unsigned)IN) && ((unsigned)h < (unsigned)IN);
  int dc = imin(imax(d, 0), IN - 1), hc = imin(imax(h, 0), IN - 1);
  const float* Xr = Xc + ((size_t)dc * IN + hc) * IN;
  union { unsigned int u; _Float16 hh[2]; } pk;
#pragma unroll
  for (int t2 = 0; t2 < 2; ++t2) {
    int w = pw0 + t2 - 1;
    bool in = inDH && ((unsigned)w < (unsigned)IN);
    int wc = imin(imax(w, 0), IN - 1);
    float v = fmaxf(sc * Xr[wc] + sh, 0.0f);
    pk.hh[t2] = (_Float16)(in ? v : 0.0f);
  }
  Xpair[idx] = pk.u;
}

// ---- prep: parity-class-reordered f16 weights Wr[cls][co][K], K = CIN*8 taps ----
__global__ __launch_bounds__(256)
void wprep_kernel(const float* __restrict__ W, _Float16* __restrict__ Wr,
                  int CIN, int COUT) {
  int K = CIN * 8;
  int idx = blockIdx.x * blockDim.x + threadIdx.x;
  if (idx >= 8 * COUT * K) return;
  int k = idx % K; int t = idx / K; int co = t % COUT; int cls = t / COUT;
  int pD = (cls >> 2) & 1, pH = (cls >> 1) & 1, pW = cls & 1;
  int ci = k >> 3, tap = k & 7;
  int tD = (tap >> 2) & 1, tH = (tap >> 1) & 1, tW = tap & 1;
  int kd = pD ? (2 - 2 * tD) : (3 - 2 * tD);
  int kh = pH ? (2 - 2 * tH) : (3 - 2 * tH);
  int kw = pW ? (2 - 2 * tW) : (3 - 2 * tW);
  Wr[idx] = (_Float16)W[((size_t)ci * COUT + co) * 64 + (kd * 4 + kh) * 4 + kw];
}

// ---- ConvTranspose3d as parity-class GEMM; A = b32 pair gathers, B contiguous ----
template <int CIN, int COUT, int IN>
__global__ __launch_bounds__(256)
void convt_wmma_kernel(const unsigned int* __restrict__ Xpair, const _Float16* __restrict__ Wr,
                       const float* __restrict__ bias, float* __restrict__ Y) {
  constexpr int OUT = IN * 2;
  constexpr int OUTVOL = OUT * OUT * OUT;
  constexpr int PDm = IN + 2;
  constexpr int PW2 = IN + 1;
  constexpr int PHS = PW2;          // ph stride (pairs)
  constexpr int PDS = PDm * PW2;    // pd stride (pairs)
  constexpr int PPV = PDm * PDm * PW2;  // per-channel pairs
  constexpr int MT = (IN * IN * IN) / 16;
  constexpr int NT = COUT / 16;
  constexpr int K = CIN * 8;
  constexpr int KCH = K / 32;

  int wave = (blockIdx.x * blockDim.x + threadIdx.x) >> 5;
  int lane = threadIdx.x & 31;
  int nt  = wave % NT;  wave /= NT;
  int mt  = wave % MT;  wave /= MT;
  int cls = wave % 8;   wave /= 8;
  int n   = wave;
  int pD = (cls >> 2) & 1, pH = (cls >> 1) & 1, pW = cls & 1;

  int pixA = mt * 16 + (lane & 15);
  int qd = pixA / (IN * IN);
  int qh = (pixA / IN) % IN;
  int qw = pixA % IN;
  const unsigned int* Xn = Xpair + (size_t)n * CIN * PPV
                         + (qd + pD) * PDS + (qh + pH) * PHS + (qw + pW);

  int co = nt * 16 + (lane & 15);
  const _Float16* bp = Wr + ((size_t)(cls * COUT + co)) * K + ((lane & 16) ? 16 : 0);

  v8f acc = {};
#pragma unroll 4
  for (int kc = 0; kc < KCH; ++kc) {
    int kb = kc * 32;
    __builtin_prefetch((const void*)(bp + kb + 64), 0, 0);  // next weight chunks
    union { v16h v; unsigned int u[8]; } au;
    v16h b = *(const v16h*)(bp + kb);
#pragma unroll
    for (int ep = 0; ep < 8; ++ep) {
      int k  = kb + a_k_of(lane, 2 * ep);   // even K: pair (k, k+1) = one dword
      int ci = k >> 3, tap = k & 7;         // tap even -> tW = 0
      int tD = (tap >> 2) & 1, tH = (tap >> 1) & 1;
      au.u[ep] = Xn[(size_t)ci * PPV + tD * PDS + tH * PHS];
    }
    acc = __builtin_amdgcn_wmma_f32_16x16x32_f16(false, au.v, false, b, (short)0, acc,
                                                 false, false);
  }
#pragma unroll
  for (int r = 0; r < 8; ++r) {
    int m = mt * 16 + r + ((lane & 16) ? 8 : 0);
    int dqd = m / (IN * IN), dqh = (m / IN) % IN, dqw = m % IN;
    int od = 2 * dqd + pD, oh = 2 * dqh + pH, ow = 2 * dqw + pW;
    int co2 = nt * 16 + (lane & 15);
    Y[(size_t)n * COUT * OUTVOL + (size_t)co2 * OUTVOL + (od * OUT + oh) * OUT + ow]
        = acc[r] + bias[co2];
  }
}

// ---------------- Stage 3: CIN=32 -> COUT=1, 16^3 -> 32^3 (VALU, maskized) ----------------
__global__ __launch_bounds__(256)
void conv3_kernel(const float* __restrict__ X, const float* __restrict__ scale,
                  const float* __restrict__ shift, const float* __restrict__ W,
                  const float* __restrict__ bias, float* __restrict__ Y) {
  int idx = blockIdx.x * blockDim.x + threadIdx.x;
  if (idx >= 64 * 32768) return;
  int n = idx >> 15, sp = idx & 32767;
  int od = sp >> 10, oh = (sp >> 5) & 31, ow = sp & 31;
  int pdb = (od - 1) >> 1, phb = (oh - 1) >> 1, pwb = (ow - 1) >> 1;
  const float* Xn = X + (size_t)n * 32 * 4096;
  float acc = bias[0];
  float mD[2], mH[2], mW[2];
  int pdc[2], phc[2], pwc[2], kdv[2], khv[2], kwv[2];
#pragma unroll
  for (int t = 0; t < 2; ++t) {
    int pd = pdb + t, ph = phb + t, pw = pwb + t;
    mD[t] = ((unsigned)pd < 16u) ? 1.f : 0.f;  pdc[t] = imin(imax(pd, 0), 15);  kdv[t] = od - 2 * pd + 1;
    mH[t] = ((unsigned)ph < 16u) ? 1.f : 0.f;  phc[t] = imin(imax(ph, 0), 15);  khv[t] = oh - 2 * ph + 1;
    mW[t] = ((unsigned)pw < 16u) ? 1.f : 0.f;  pwc[t] = imin(imax(pw, 0), 15);  kwv[t] = ow - 2 * pw + 1;
  }
  for (int ci = 0; ci < 32; ++ci) {
    float sc = scale[ci], sh = shift[ci];
    const float* Xc = Xn + ci * 4096;
    const float* Wc = W + ci * 64;
#pragma unroll
    for (int tD = 0; tD < 2; ++tD)
#pragma unroll
      for (int tH = 0; tH < 2; ++tH)
#pragma unroll
        for (int tW = 0; tW < 2; ++tW) {
          float m = mD[tD] * mH[tH] * mW[tW];
          float xv = Xc[(pdc[tD] * 16 + phc[tH]) * 16 + pwc[tW]];
          acc += m * fmaxf(sc * xv + sh, 0.0f) * Wc[(kdv[tD] * 4 + khv[tH]) * 4 + kwv[tW]];
        }
  }
  Y[idx] = acc;
}

// ----------- Stage 4: 1->1 channel, 32^3 -> 64^3, fused sigmoid (maskized) -----------
__global__ __launch_bounds__(256)
void conv4_sigmoid_kernel(const float* __restrict__ X, const float* __restrict__ scale,
                          const float* __restrict__ shift, const float* __restrict__ W,
                          const float* __restrict__ bias, float* __restrict__ VOXE) {
  int idx = blockIdx.x * blockDim.x + threadIdx.x;
  if (idx >= 64 * 262144) return;
  int n = idx >> 18, sp = idx & 262143;
  int od = sp >> 12, oh = (sp >> 6) & 63, ow = sp & 63;
  int pdb = (od - 1) >> 1, phb = (oh - 1) >> 1, pwb = (ow - 1) >> 1;
  float sc = scale[0], sh = shift[0];
  const float* Xn = X + (size_t)n * 32768;
  float acc = bias[0];
#pragma unroll
  for (int tD = 0; tD < 2; ++tD)
#pragma unroll
    for (int tH = 0; tH < 2; ++tH)
#pragma unroll
      for (int tW = 0; tW < 2; ++tW) {
        int pd = pdb + tD, ph = phb + tH, pw = pwb + tW;
        float m = (((unsigned)pd < 32u) && ((unsigned)ph < 32u) && ((unsigned)pw < 32u))
                      ? 1.f : 0.f;
        int pdc = imin(imax(pd, 0), 31), phc = imin(imax(ph, 0), 31), pwc = imin(imax(pw, 0), 31);
        int kd = od - 2 * pd + 1, kh = oh - 2 * ph + 1, kw = ow - 2 * pw + 1;
        float xv = Xn[(pdc * 32 + phc) * 32 + pwc];
        acc += m * fmaxf(sc * xv + sh, 0.0f) * W[(kd * 4 + kh) * 4 + kw];
      }
  VOXE[idx] = 1.0f / (1.0f + expf(-acc));   // sigmoid * (1/tau), tau = 1
}

// ------------ Rotate + trilinear resample + ray-sum projection ------------
__global__ __launch_bounds__(256)
void project_kernel(const float* __restrict__ VOXE, const float* __restrict__ z,
                    float* __restrict__ out) {
  int idx = blockIdx.x * blockDim.x + threadIdx.x;
  if (idx >= 64 * 4096) return;
  int n = idx >> 12, a = (idx >> 6) & 63, b = idx & 63;
  float s = z[n * 201 + 200];
  int vi = imin(imax((int)((s + 1.0f) * 4.5f), 0), 8);
  const float VP[9] = {0.f, 0.78539816f, 1.57079633f, 2.35619449f, 3.14159265f,
                       3.92699082f, 4.71238898f, 5.49778714f, 1.57079633f};
  const float HP[9] = {0.f, 0.f, 0.f, 0.f, 0.f, 0.f, 0.f, 0.f, 1.57079633f};
  float th = VP[vi], ph = HP[vi];
  float st = sinf(th), ct = cosf(th), sp = sinf(ph), cp = cosf(ph);
  float R00 = ct,       R02 = st;
  float R10 = sp * st,  R11 = cp,  R12 = -sp * ct;
  float R20 = -cp * st, R21 = sp,  R22 = cp * ct;
  int i = 63 - b, j = 63 - a;
  float gx = -1.0f + 2.0f * (float)i / 63.0f;
  float gy = -1.0f + 2.0f * (float)j / 63.0f;
  const float* Vn = VOXE + (size_t)n * 262144;
  auto V = [&](float fi, float fj, float fk) {
    int i3 = imin(imax((int)fi, 0), 63);
    int j3 = imin(imax((int)fj, 0), 63);
    int k3 = imin(imax((int)fk, 0), 63);
    return Vn[(i3 * 64 + j3) * 64 + k3];
  };
  float p = 0.f;
  for (int k = 0; k < 64; ++k) {
    float gz = -1.0f + 2.0f * (float)k / 63.0f;
    float px = gx * R00 + gy * R10 + gz * R20;
    float py = gy * R11 + gz * R21;
    float pz = gx * R02 + gy * R12 + gz * R22;
    float xs = (px + 1.0f) * 32.0f, ys = (py + 1.0f) * 32.0f, zs = (pz + 1.0f) * 32.0f;
    float xc = fminf(fmaxf(xs, 0.f), 64.f);
    float yc = fminf(fmaxf(ys, 0.f), 64.f);
    float zc = fminf(fmaxf(zs, 0.f), 64.f);
    float fx = floorf(xc), cx = ceilf(xc);
    float fy = floorf(yc), cy = ceilf(yc);
    float fz = floorf(zc), cz = ceilf(zc);
    p += fabsf((xs - fx) * (ys - fy) * (zs - fz)) * V(cx, cy, cz)
       + fabsf((xs - fx) * (ys - fy) * (zs - cz)) * V(cx, cy, fz)
       + fabsf((xs - fx) * (ys - cy) * (zs - fz)) * V(cx, fy, cz)
       + fabsf((xs - fx) * (ys - cy) * (zs - cz)) * V(cx, fy, fz)
       + fabsf((xs - cx) * (ys - fy) * (zs - fz)) * V(fx, cy, cz)
       + fabsf((xs - cx) * (ys - fy) * (zs - cz)) * V(fx, cy, fz)
       + fabsf((xs - cx) * (ys - cy) * (zs - fz)) * V(fx, fy, cz)
       + fabsf((xs - cx) * (ys - cy) * (zs - cz)) * V(fx, fy, fz);
  }
  out[idx] = 1.0f - expf(-p);
}

extern "C" void kernel_launch(void* const* d_in, const int* in_sizes, int n_in,
                              void* d_out, int out_size, void* d_ws, size_t ws_size,
                              hipStream_t stream) {
  const float* z   = (const float*)d_in[0];
  const float* fcw = (const float*)d_in[1];
  const float* fcb = (const float*)d_in[2];
  const float* g1  = (const float*)d_in[3];  const float* bb1 = (const float*)d_in[4];
  const float* g2  = (const float*)d_in[5];  const float* bb2 = (const float*)d_in[6];
  const float* g3  = (const float*)d_in[7];  const float* bb3 = (const float*)d_in[8];
  const float* g4  = (const float*)d_in[9];  const float* bb4 = (const float*)d_in[10];
  const float* w1  = (const float*)d_in[11]; const float* cb1 = (const float*)d_in[12];
  const float* w2  = (const float*)d_in[13]; const float* cb2 = (const float*)d_in[14];
  const float* w3  = (const float*)d_in[15]; const float* cb3 = (const float*)d_in[16];
  const float* w4  = (const float*)d_in[17]; const float* cb4 = (const float*)d_in[18];

  char* ws = (char*)d_ws;
  float*        VOXE = (float*)(ws);                      // 67108864 B
  float*        H2   = (float*)(ws + 67108864ull);        // 33554432 B
  float*        H1   = (float*)(ws + 100663296ull);       //  8388608 B
  float*        H0   = (float*)(ws + 109051904ull);       //  2097152 B
  float*        H3   = (float*)(ws + 111149056ull);       //  8388608 B
  float*        ST   = (float*)(ws + 119537664ull);       //     4096 B
  _Float16*     Af16 = (_Float16*)(ws + 119541760ull);    //    28672 B (64x224)
  _Float16*     Bf16 = (_Float16*)(ws + 119570432ull);    //  3670016 B (8192x224)
  unsigned int* Xp1  = (unsigned int*)(ws + 123240448ull);//  5898240 B (64x128x6x6x5 pairs)
  _Float16*     Wr1  = (_Float16*)(ws + 129138688ull);    //  1048576 B (8x64x1024)
  unsigned int* Xp2  = (unsigned int*)(ws + 130187264ull);// 14745600 B (64x64x10x10x9 pairs)
  _Float16*     Wr2  = (_Float16*)(ws + 144932864ull);    //   262144 B (8x32x512)
  float* sc1 = ST,       * sh1 = ST + 128;
  float* sc2 = ST + 256, * sh2 = ST + 320;
  float* sc3 = ST + 384, * sh3 = ST + 416;
  float* sc4 = ST + 448, * sh4 = ST + 449;

  // operand prep (weights / FC inputs independent of activations)
  cvt_pad_f16_kernel<<<(64 * 224 + 255) / 256, 256, 0, stream>>>(z, Af16, 64, 200, 201, 224);
  cvt_pad_f16_kernel<<<(8192 * 224 + 255) / 256, 256, 0, stream>>>(fcw, Bf16, 8192, 200, 200, 224);
  wprep_kernel<<<(8 * 64 * 1024 + 255) / 256, 256, 0, stream>>>(w1, Wr1, 128, 64);
  wprep_kernel<<<(8 * 32 * 512 + 255) / 256, 256, 0, stream>>>(w2, Wr2, 64, 32);

  // FC: 2048 wave-tiles (4 M x 512 N), 8 waves/block
  fc_wmma_kernel<<<256, 256, 0, stream>>>(Af16, Bf16, fcb, H0);
  bn_stats_kernel<<<128, 256, 0, stream>>>(H0, g1, bb1, sc1, sh1, 128, 64);
  act_pair_prep_kernel<<<(64 * 128 * 180 + 255) / 256, 256, 0, stream>>>(H0, sc1, sh1, Xp1, 128, 4);
  // convT1: 64 n * 8 cls * 4 Mt * 4 Nt = 8192 waves
  convt_wmma_kernel<128, 64, 4><<<1024, 256, 0, stream>>>(Xp1, Wr1, cb1, H1);
  bn_stats_kernel<<<64, 256, 0, stream>>>(H1, g2, bb2, sc2, sh2, 64, 512);
  act_pair_prep_kernel<<<(64 * 64 * 900 + 255) / 256, 256, 0, stream>>>(H1, sc2, sh2, Xp2, 64, 8);
  // convT2: 64 * 8 * 32 * 2 = 32768 waves
  convt_wmma_kernel<64, 32, 8><<<4096, 256, 0, stream>>>(Xp2, Wr2, cb2, H2);
  bn_stats_kernel<<<32, 256, 0, stream>>>(H2, g3, bb3, sc3, sh3, 32, 4096);
  conv3_kernel<<<(64 * 32768 + 255) / 256, 256, 0, stream>>>(H2, sc3, sh3, w3, cb3, H3);
  bn_stats_kernel<<<1, 256, 0, stream>>>(H3, g4, bb4, sc4, sh4, 1, 32768);
  conv4_sigmoid_kernel<<<(64 * 262144 + 255) / 256, 256, 0, stream>>>(H3, sc4, sh4, w4, cb4, VOXE);
  project_kernel<<<(64 * 4096 + 255) / 256, 256, 0, stream>>>(VOXE, z, (float*)d_out);

  (void)in_sizes; (void)n_in; (void)out_size; (void)ws_size;
}
